// SoftmaxTree_75256416961192
// MI455X (gfx1250) — matrix operand order repaired and checked by reading, not compiled, not run on hardware
//
#include <hip/hip_runtime.h>
#include <hip/hip_bf16.h>
#include <math.h>

// ---------------------------------------------------------------------------
// SoftmaxTree for MI455X (gfx1250, wave32, WMMA)
//   logits_i = x @ W_i^T + b_i   (6 GEMMs, bf16 WMMA, f32 accumulate)
//   level 0: row softmax; levels 1..5: per-parent column-segment softmax
//            multiplied by parent's probability (in place in d_out).
// ---------------------------------------------------------------------------

#define DIM 768
#define NB  2048

typedef __attribute__((ext_vector_type(16))) __bf16 v16bf;
typedef __attribute__((ext_vector_type(8)))  __bf16 v8bf;
typedef __attribute__((ext_vector_type(4)))  __bf16 v4bf;
typedef __attribute__((ext_vector_type(8)))  float  v8f;

__device__ __forceinline__ v4bf cvt4(const float4 v) {
  // native fptrunc f32->bf16; backend should select packed bf16 converts
  v4bf r = { (__bf16)v.x, (__bf16)v.y, (__bf16)v.z, (__bf16)v.w };
  return r;
}

#define KT   32     // K per WMMA step (bf16)
#define ASTR 40     // LDS row stride in bf16 elems (32 + 8 pad -> 80B rows)
#define TM   128    // block tile M
#define TN   64     // block tile N

__global__ __launch_bounds__(256)
void wmma_logits_kernel(const float* __restrict__ X,
                        const float* __restrict__ W,
                        const float* __restrict__ bias,
                        float* __restrict__ out, int C) {
  __shared__ __attribute__((aligned(16))) __bf16 sA[2][TM * ASTR];
  __shared__ __attribute__((aligned(16))) __bf16 sB[2][TN * ASTR];

  const int tid    = threadIdx.x;
  const int blockM = blockIdx.y * TM;
  const int blockN = blockIdx.x * TN;

  // ---- staging indices: 256 threads, float4 granularity ----
  const int stRow = tid >> 3;          // 0..31
  const int stCol = (tid & 7) << 2;    // 0,4,...,28

  // software-pipeline registers (next tile, f32)
  float4 ra[4];
  float4 rb[2];

  auto gload = [&](int kt) {           // issue global loads only (no wait here)
    const int k0 = kt * KT;
#pragma unroll
    for (int p = 0; p < 4; ++p)
      ra[p] = *(const float4*)(X + (size_t)(blockM + stRow + p * 32) * DIM + k0 + stCol);
#pragma unroll
    for (int p = 0; p < 2; ++p) {
      const int n = blockN + stRow + p * 32;
      rb[p] = (n < C) ? *(const float4*)(W + (size_t)n * DIM + k0 + stCol)
                      : make_float4(0.f, 0.f, 0.f, 0.f);
    }
  };
  auto lstore = [&](int buf) {         // convert + LDS store (consumes ra/rb)
#pragma unroll
    for (int p = 0; p < 4; ++p)
      *(v4bf*)&sA[buf][(stRow + p * 32) * ASTR + stCol] = cvt4(ra[p]);
#pragma unroll
    for (int p = 0; p < 2; ++p)
      *(v4bf*)&sB[buf][(stRow + p * 32) * ASTR + stCol] = cvt4(rb[p]);
  };

  // ---- wave tiling: 8 waves = 4(M) x 2(N), 32x32 per wave ----
  const int wave  = tid >> 5;
  const int lane  = tid & 31;
  const int waveM = (wave & 3) * 32;
  const int waveN = (wave >> 2) * 32;
  const int lh    = lane >> 4;         // lane half (0/1)
  const int lm    = lane & 15;

  v8f acc[2][2] = {};

  gload(0);
  lstore(0);
  __syncthreads();

  const int NT = DIM / KT;             // 24 K-steps
  for (int kt = 0; kt < NT; ++kt) {
    const int cur = kt & 1;
    if (kt + 1 < NT) gload(kt + 1);    // global loads in flight under WMMA

    // A fragments: 16x32 bf16, ISA layout: lane half h holds
    //   K in [8h, 8h+8) (frag[0..7]) and [16+8h, 16+8h+8) (frag[8..15])
    v16bf af[2];
#pragma unroll
    for (int mi = 0; mi < 2; ++mi) {
      const __bf16* base = &sA[cur][(waveM + mi * 16 + lm) * ASTR];
      v8bf lo = *(const v8bf*)(base + 8 * lh);
      v8bf hi = *(const v8bf*)(base + 16 + 8 * lh);
#pragma unroll
      for (int i = 0; i < 8; ++i) { af[mi][i] = lo[i]; af[mi][i + 8] = hi[i]; }
    }
    // B fragments: 32x16 bf16; lane holds column n=lm, K in [8h, 8h+8) and
    // [16+8h, 16+8h+8), matching A (slices of a W row, NT gemm, no transpose).
    v16bf bfr[2];
#pragma unroll
    for (int ni = 0; ni < 2; ++ni) {
      const __bf16* base = &sB[cur][(waveN + ni * 16 + lm) * ASTR];
      v8bf lo = *(const v8bf*)(base + 8 * lh);
      v8bf hi = *(const v8bf*)(base + 16 + 8 * lh);
#pragma unroll
      for (int i = 0; i < 8; ++i) { bfr[ni][i] = lo[i]; bfr[ni][i + 8] = hi[i]; }
    }

#pragma unroll
    for (int mi = 0; mi < 2; ++mi)
#pragma unroll
      for (int ni = 0; ni < 2; ++ni)
        acc[mi][ni] = __builtin_amdgcn_wmma_f32_16x16x32_bf16(
            false, af[mi], false, bfr[ni], (short)0, acc[mi][ni], false, false);

    if (kt + 1 < NT) lstore(cur ^ 1);  // waits loads, converts, fills other buf
    __syncthreads();
  }

  // ---- epilogue: bias add + store (C/D layout: VGPR r -> M = r + 8*lh) ----
#pragma unroll
  for (int mi = 0; mi < 2; ++mi) {
#pragma unroll
    for (int ni = 0; ni < 2; ++ni) {
      const int col = blockN + waveN + ni * 16 + lm;
      if (col >= C) continue;
      const int r0 = blockM + waveM + mi * 16 + 8 * lh;
      const float bv = bias[col];
#pragma unroll
      for (int r = 0; r < 8; ++r)
        out[(size_t)(r0 + r) * C + col] = acc[mi][ni][r] + bv;
    }
  }
}

// ---- level-0: plain softmax over C columns of each row, in place ----
__global__ void softmax_rows_kernel(float* __restrict__ o, int C) {
  const int row = blockIdx.x * blockDim.x + threadIdx.x;
  if (row >= NB) return;
  float* p = o + (size_t)row * C;
  float mx = -INFINITY;
  for (int c = 0; c < C; ++c) mx = fmaxf(mx, p[c]);
  float s = 0.f;
  for (int c = 0; c < C; ++c) s += __expf(p[c] - mx);
  const float inv = 1.f / s;
  for (int c = 0; c < C; ++c) p[c] = __expf(p[c] - mx) * inv;
}

// ---- levels 1..5: segment softmax over sibling column groups, times the
//      parent's probability. One thread per (row, parent); sibling groups are
//      contiguous because the parent array is sorted (binary search range). ----
__global__ void seg_softmax_kernel(float* __restrict__ cur,
                                   const float* __restrict__ parent,
                                   const int* __restrict__ par,
                                   int C, int P) {
  const int pid = blockIdx.x * blockDim.x + threadIdx.x;
  const int row = blockIdx.y;
  if (pid >= P) return;
  // lower_bound(par, pid)
  int lo = 0, hi = C;
  while (lo < hi) { int m = (lo + hi) >> 1; if (par[m] < pid) lo = m + 1; else hi = m; }
  const int s = lo;
  hi = C;                                   // upper_bound(par, pid)
  while (lo < hi) { int m = (lo + hi) >> 1; if (par[m] <= pid) lo = m + 1; else hi = m; }
  const int e = lo;
  if (s >= e) return;

  float* lp = cur + (size_t)row * C;
  float mx = -INFINITY;
  for (int i = s; i < e; ++i) mx = fmaxf(mx, lp[i]);
  float sum = 0.f;
  for (int i = s; i < e; ++i) sum += __expf(lp[i] - mx);
  const float pscale = parent[(size_t)row * P + pid] / sum;
  for (int i = s; i < e; ++i) lp[i] = __expf(lp[i] - mx) * pscale;
}

// ---------------------------------------------------------------------------
extern "C" void kernel_launch(void* const* d_in, const int* in_sizes, int n_in,
                              void* d_out, int out_size, void* d_ws, size_t ws_size,
                              hipStream_t stream) {
  static const int CLS[6] = {20, 70, 250, 800, 3000, 12000};

  const float* x = (const float*)d_in[0];
  const float* Wv[6]; const float* bv[6];
  for (int i = 0; i < 6; ++i) {
    Wv[i] = (const float*)d_in[1 + 2 * i];
    bv[i] = (const float*)d_in[2 + 2 * i];
  }
  const int* pv[5];
  for (int l = 0; l < 5; ++l) pv[l] = (const int*)d_in[13 + l];

  float* out = (float*)d_out;
  size_t off[7]; off[0] = 0;
  for (int i = 0; i < 6; ++i) off[i + 1] = off[i] + (size_t)NB * CLS[i];

  // 1) all six GEMMs -> logits written into their d_out regions
  for (int lvl = 0; lvl < 6; ++lvl) {
    dim3 grid((CLS[lvl] + TN - 1) / TN, NB / TM);
    wmma_logits_kernel<<<grid, 256, 0, stream>>>(x, Wv[lvl], bv[lvl],
                                                 out + off[lvl], CLS[lvl]);
  }
  // 2) level-0 softmax in place
  softmax_rows_kernel<<<(NB + 255) / 256, 256, 0, stream>>>(out, CLS[0]);
  // 3) levels 1..5: segment softmax * gathered parent prob, in place
  for (int lvl = 1; lvl < 6; ++lvl) {
    dim3 g((CLS[lvl - 1] + 255) / 256, NB);
    seg_softmax_kernel<<<g, 256, 0, stream>>>(out + off[lvl], out + off[lvl - 1],
                                              pv[lvl - 1], CLS[lvl], CLS[lvl - 1]);
  }
}
